// HAN_87514253623570
// MI455X (gfx1250) — matrix operand order repaired and checked by reading, not compiled, not run on hardware
//
#include <hip/hip_runtime.h>
#include <hip/hip_bf16.h>
#include <math.h>

typedef __attribute__((ext_vector_type(16))) __bf16 v16bf;
typedef __attribute__((ext_vector_type(8)))  __bf16 v8bf;
typedef __attribute__((ext_vector_type(8)))  float  v8f;

#define HAN_N     100000
#define HAN_E     1600000
#define HAN_DIN   256
#define HAN_DH    128
#define HAN_H     4
#define HAN_F     32
#define HAN_C     8
#define HAN_LDS_PAD 8   // +8 bf16 per row: 528B row stride -> conflict-free column reads

// ---------------------------------------------------------------- zero scratch
__global__ void han_zero(float* __restrict__ p, long long count) {
    long long i = (long long)blockIdx.x * blockDim.x + threadIdx.x;
    long long stride = (long long)gridDim.x * blockDim.x;
    for (; i < count; i += stride) p[i] = 0.0f;
}

// ---------------------------------------------------------------- f32 -> bf16 (flat)
__global__ void han_cvt(const float* __restrict__ src, __bf16* __restrict__ dst,
                        long long count) {
    long long i = (long long)blockIdx.x * blockDim.x + threadIdx.x;
    if (i >= count) return;
    dst[i] = (__bf16)src[i];
}

// --------------------------------------------- fp32 [rows x cols] -> bf16 [cols x rows]
__global__ void han_cvt_transpose(const float* __restrict__ src, __bf16* __restrict__ dst,
                                  int rows, int cols) {
    int i = blockIdx.x * blockDim.x + threadIdx.x;
    if (i >= rows * cols) return;
    int r = i / cols, c = i - r * cols;
    dst[(size_t)c * rows + r] = (__bf16)src[i];
}

// ---------------------------------------------------------------- WMMA GEMM (16 x 128 strip per wave)
// C[M x 128] = A[M x K](bf16) * Bt[128 x K](bf16, pre-transposed), fp32 accumulate.
// Bt is staged in LDS once per workgroup (row-padded, bank-conflict-free reads).
// Epilogues:
//   red!=null, do_tanh==0: el[row*H + head] += sum_col acc*attn[col]   (per-row head dots)
//   red!=null, do_tanh==1: red[0] += sum_all tanh(acc+bias[col])*attn[col] (global scalar)
__global__ void han_wmma_gemm(const __bf16* __restrict__ A, int K,
                              const __bf16* __restrict__ Bt,
                              float* __restrict__ C,
                              const float* __restrict__ attn,
                              float* __restrict__ red,
                              const float* __restrict__ bias,
                              int do_tanh, int red_per_row,
                              int M) {
    extern __shared__ __bf16 blds[];               // [128 x (K + HAN_LDS_PAD)]
    const int ldsK = K + HAN_LDS_PAD;

    // ---- cooperative Bt -> LDS stage (ALL waves participate, then barrier) ----
    const int nchunk = K >> 3;                     // v8bf chunks per row
    for (int idx = threadIdx.x; idx < HAN_DH * nchunk; idx += blockDim.x) {
        int r = idx / nchunk;
        int c = (idx - r * nchunk) << 3;
        *(v8bf*)&blds[r * ldsK + c] = *(const v8bf*)&Bt[(size_t)r * K + c];
    }
    __syncthreads();

    int t = blockIdx.x * blockDim.x + threadIdx.x;
    int wave = t >> 5;
    if (wave >= (M >> 4)) return;                  // whole-wave early out AFTER barrier

    int lane = threadIdx.x & 31;
    int half = lane >> 4;                          // which 16-lane half
    int l16  = lane & 15;
    int hi   = half << 4;                          // A K-chunk offset: 0 or 16

    // 32-bit lane offsets from uniform bases (saddr + voffset addressing)
    const int aoff = (wave * 16 + l16) * K + hi;   // A: M = lane&15
    const int boff = l16 * ldsK + (half << 4);     // B: N = lane&15, K = half*16 + ...

    v8f acc[8] = {};
    for (int k0 = 0; k0 < K; k0 += 32) {
        __builtin_prefetch(A + aoff + k0 + 64, 0, 1);      // global_prefetch_b8
        v8bf c0 = *(const v8bf*)(A + aoff + k0);           // K = k0 + hi + (0..7)
        v8bf c1 = *(const v8bf*)(A + aoff + k0 + 8);       // K = k0 + hi + 8 + (0..7)
        v16bf av;
        #pragma unroll
        for (int i = 0; i < 8; ++i) { av[i] = c0[i]; av[8 + i] = c1[i]; }
        #pragma unroll
        for (int ct = 0; ct < 8; ++ct) {                   // 8 column tiles = full 128 cols
            v16bf bv = *(const v16bf*)&blds[boff + ct * 16 * ldsK + k0];
            acc[ct] = __builtin_amdgcn_wmma_f32_16x16x32_bf16(false, av, false, bv,
                                                              (short)0, acc[ct], false, false);
        }
    }

    if (C) {
        #pragma unroll
        for (int ct = 0; ct < 8; ++ct)
            #pragma unroll
            for (int v = 0; v < 8; ++v) {
                int row = wave * 16 + v + (half << 3);     // D: M = v + 8*half
                C[(size_t)row * HAN_DH + ct * 16 + l16] = acc[ct][v];
            }
    }
    if (red) {
        if (red_per_row) {                                  // GAT el/er head dots
            #pragma unroll
            for (int ct = 0; ct < 8; ++ct) {
                float av = attn[ct * 16 + l16];
                int head = ct >> 1;                         // F=32 -> 2 tiles per head
                #pragma unroll
                for (int v = 0; v < 8; ++v) {
                    float p = acc[ct][v] * av;
                    #pragma unroll
                    for (int off = 1; off < 16; off <<= 1) p += __shfl_xor(p, off, 32);
                    if (l16 == 0) {
                        int row = wave * 16 + v + (half << 3);
                        atomicAdd(&red[(size_t)row * red_per_row + head], p);
                    }
                }
            }
        } else {                                            // semantic: global scalar
            float local = 0.0f;
            #pragma unroll
            for (int ct = 0; ct < 8; ++ct) {
                float av = attn[ct * 16 + l16];
                float bv = bias[ct * 16 + l16];
                #pragma unroll
                for (int v = 0; v < 8; ++v)
                    local += (do_tanh ? tanhf(acc[ct][v] + bv) : (acc[ct][v] + bv)) * av;
            }
            #pragma unroll
            for (int off = 16; off >= 1; off >>= 1) local += __shfl_xor(local, off, 32);
            if (lane == 0) atomicAdd(red, local);
        }
    }
}

// ---------------------------------------------------------------- per-edge attention + scatter
__global__ void han_edge(const int* __restrict__ src, const int* __restrict__ dst,
                         const float* __restrict__ ew,
                         const float* __restrict__ fs,
                         const float* __restrict__ el, const float* __restrict__ er,
                         float* __restrict__ agg, int nedges) {
    int t = blockIdx.x * blockDim.x + threadIdx.x;
    if (t >= nedges * 32) return;
    int e    = t >> 5;
    int lane = t & 31;
    int s = src[e], d = dst[e];
    int h = lane >> 3;                       // head for cols [lane*4, lane*4+4)
    float ee = el[s * HAN_H + h] + er[d * HAN_H + h];
    float le = ee > 0.0f ? ee : 0.2f * ee;   // leaky_relu(0.2)
    float a  = ew[e] / (1.0f + expf(-le));   // sigmoid * edge_weight
    const float4 v = *(const float4*)(fs + (size_t)s * HAN_DH + lane * 4);
    float* o = agg + (size_t)d * HAN_DH + lane * 4;
    atomicAdd(o + 0, v.x * a);
    atomicAdd(o + 1, v.y * a);
    atomicAdd(o + 2, v.z * a);
    atomicAdd(o + 3, v.w * a);
}

// ---------------------------------------------------------------- bias + ELU (+ bf16 copy)
__global__ void han_bias_elu(const float* __restrict__ agg, const float* __restrict__ b,
                             float* __restrict__ z, __bf16* __restrict__ zb, int total) {
    int i = blockIdx.x * blockDim.x + threadIdx.x;
    if (i >= total) return;
    float v = agg[i] + b[i & (HAN_DH - 1)];
    float r = v > 0.0f ? v : expm1f(v);
    z[i]  = r;
    zb[i] = (__bf16)r;
}

// ---------------------------------------------------------------- 2-way softmax of means
__global__ void han_beta(const float* __restrict__ ssum, float* __restrict__ beta, float invN) {
    float w0 = ssum[0] * invN, w1 = ssum[1] * invN;
    float m = fmaxf(w0, w1);
    float e0 = expf(w0 - m), e1 = expf(w1 - m);
    float inv = 1.0f / (e0 + e1);
    beta[0] = e0 * inv;
    beta[1] = e1 * inv;
}

// ---------------------------------------------------------------- beta-combine + 128x8 head
__global__ void han_final(const float* __restrict__ z0, const float* __restrict__ z1,
                          const float* __restrict__ beta,
                          const float* __restrict__ Wo, const float* __restrict__ bo,
                          float* __restrict__ out, int n) {
    int t = blockIdx.x * blockDim.x + threadIdx.x;
    if (t >= n * 32) return;
    int node = t >> 5;
    int lane = t & 31;
    float b0 = beta[0], b1 = beta[1];
    const float4 p0 = *(const float4*)(z0 + (size_t)node * HAN_DH + lane * 4);
    const float4 p1 = *(const float4*)(z1 + (size_t)node * HAN_DH + lane * 4);
    float h[4] = { b0*p0.x + b1*p1.x, b0*p0.y + b1*p1.y,
                   b0*p0.z + b1*p1.z, b0*p0.w + b1*p1.w };
    float acc[HAN_C];
    #pragma unroll
    for (int c = 0; c < HAN_C; ++c) {
        float s = 0.0f;
        #pragma unroll
        for (int j = 0; j < 4; ++j) s += h[j] * Wo[(lane * 4 + j) * HAN_C + c];
        acc[c] = s;
    }
    #pragma unroll
    for (int off = 16; off >= 1; off >>= 1)
        #pragma unroll
        for (int c = 0; c < HAN_C; ++c) acc[c] += __shfl_xor(acc[c], off, 32);
    if (lane == 0) {
        #pragma unroll
        for (int c = 0; c < HAN_C; ++c) out[(size_t)node * HAN_C + c] = acc[c] + bo[c];
    }
}

extern "C" void kernel_launch(void* const* d_in, const int* in_sizes, int n_in,
                              void* d_out, int out_size, void* d_ws, size_t ws_size,
                              hipStream_t stream) {
    const float* x    = (const float*)d_in[0];
    const int*   src0 = (const int*)  d_in[1];
    const int*   dst0 = (const int*)  d_in[2];
    const float* ew0  = (const float*)d_in[3];
    const int*   src1 = (const int*)  d_in[4];
    const int*   dst1 = (const int*)  d_in[5];
    const float* ew1  = (const float*)d_in[6];
    const float* Wsrc0 = (const float*)d_in[7];
    const float* Wdst0 = (const float*)d_in[8];
    const float* al0   = (const float*)d_in[9];
    const float* ar0   = (const float*)d_in[10];
    const float* b0    = (const float*)d_in[11];
    const float* Wsrc1 = (const float*)d_in[12];
    const float* Wdst1 = (const float*)d_in[13];
    const float* al1   = (const float*)d_in[14];
    const float* ar1   = (const float*)d_in[15];
    const float* b1    = (const float*)d_in[16];
    const float* Wp1   = (const float*)d_in[17];
    const float* bp1   = (const float*)d_in[18];
    const float* Wp2   = (const float*)d_in[19];
    const float* Wo    = (const float*)d_in[20];
    const float* bo    = (const float*)d_in[21];
    float* out = (float*)d_out;

    const int N = HAN_N, E = HAN_E;
    const size_t NF = (size_t)N * HAN_DH;

    // ---- workspace layout (fp32 region, then bf16 region) ----
    float* w    = (float*)d_ws;
    float* fs0  = w;
    float* fs1  = w + 1 * NF;
    float* z0   = w + 2 * NF;
    float* z1   = w + 3 * NF;
    float* agg0 = w + 4 * NF;      // zeroed region starts here
    float* agg1 = w + 5 * NF;
    float* el0  = w + 6 * NF;
    float* er0  = el0 + (size_t)N * HAN_H;
    float* el1  = er0 + (size_t)N * HAN_H;
    float* er1  = el1 + (size_t)N * HAN_H;
    float* ssum = er1 + (size_t)N * HAN_H;   // 2 floats
    float* beta = ssum + 2;                  // 2 floats
    size_t f32_count = 6 * NF + 4 * (size_t)N * HAN_H + 4;
    __bf16* bb   = (__bf16*)(w + ((f32_count + 63) / 64) * 64);
    __bf16* wt0s = bb;                                // [128 x 256] each
    __bf16* wt0d = wt0s + HAN_DH * HAN_DIN;
    __bf16* wt1s = wt0d + HAN_DH * HAN_DIN;
    __bf16* wt1d = wt1s + HAN_DH * HAN_DIN;
    __bf16* wp1t = wt1d + HAN_DH * HAN_DIN;           // [128 x 128]
    __bf16* xb   = wp1t + HAN_DH * HAN_DH;            // [N x 256]
    __bf16* z0b  = xb + (size_t)N * HAN_DIN;          // [N x 128]
    __bf16* z1b  = z0b + NF;                          // [N x 128]

    // 1) zero accumulators (agg0, agg1, el/er, ssum) — every call, graph-replay safe
    long long zcount = 2 * (long long)NF + 4LL * N * HAN_H + 2;
    han_zero<<<4096, 256, 0, stream>>>(agg0, zcount);

    // 2) x -> bf16 (amortized over 4 GEMMs); weight transpose + bf16 convert
    {
        long long xcount = (long long)N * HAN_DIN;
        han_cvt<<<(int)((xcount + 255) / 256), 256, 0, stream>>>(x, xb, xcount);
        int tot = HAN_DIN * HAN_DH, blk = (tot + 255) / 256;
        han_cvt_transpose<<<blk, 256, 0, stream>>>(Wsrc0, wt0s, HAN_DIN, HAN_DH);
        han_cvt_transpose<<<blk, 256, 0, stream>>>(Wdst0, wt0d, HAN_DIN, HAN_DH);
        han_cvt_transpose<<<blk, 256, 0, stream>>>(Wsrc1, wt1s, HAN_DIN, HAN_DH);
        han_cvt_transpose<<<blk, 256, 0, stream>>>(Wdst1, wt1d, HAN_DIN, HAN_DH);
        int tot2 = HAN_DH * HAN_DH, blk2 = (tot2 + 255) / 256;
        han_cvt_transpose<<<blk2, 256, 0, stream>>>(Wp1, wp1t, HAN_DH, HAN_DH);
    }

    // 3) feature GEMMs: fs (stored, + el dot), fd (not stored, er dot only)
    {
        int blocks = ((N / 16) * 32 + 255) / 256;
        size_t shm = (size_t)(HAN_DIN + HAN_LDS_PAD) * HAN_DH * sizeof(__bf16); // 67.5 KB
        han_wmma_gemm<<<blocks, 256, shm, stream>>>(xb, HAN_DIN, wt0s, fs0, al0, el0,
                                                    nullptr, 0, HAN_H, N);
        han_wmma_gemm<<<blocks, 256, shm, stream>>>(xb, HAN_DIN, wt0d, nullptr, ar0, er0,
                                                    nullptr, 0, HAN_H, N);
        han_wmma_gemm<<<blocks, 256, shm, stream>>>(xb, HAN_DIN, wt1s, fs1, al1, el1,
                                                    nullptr, 0, HAN_H, N);
        han_wmma_gemm<<<blocks, 256, shm, stream>>>(xb, HAN_DIN, wt1d, nullptr, ar1, er1,
                                                    nullptr, 0, HAN_H, N);
    }

    // 4) edge attention + scatter-add (bandwidth-bound stage)
    {
        int blocks = (E * 32 + 255) / 256;
        han_edge<<<blocks, 256, 0, stream>>>(src0, dst0, ew0, fs0, el0, er0, agg0, E);
        han_edge<<<blocks, 256, 0, stream>>>(src1, dst1, ew1, fs1, el1, er1, agg1, E);
    }

    // 5) bias + ELU -> z0, z1 (f32 for final combine, bf16 for semantic GEMM)
    {
        int tot = (int)NF, blocks = (tot + 255) / 256;
        han_bias_elu<<<blocks, 256, 0, stream>>>(agg0, b0, z0, z0b, tot);
        han_bias_elu<<<blocks, 256, 0, stream>>>(agg1, b1, z1, z1b, tot);
    }

    // 6) semantic attention score GEMMs (tanh epilogue, scalar reduction; no C store)
    {
        int blocks = ((N / 16) * 32 + 255) / 256;
        size_t shm = (size_t)(HAN_DH + HAN_LDS_PAD) * HAN_DH * sizeof(__bf16);  // 34 KB
        han_wmma_gemm<<<blocks, 256, shm, stream>>>(z0b, HAN_DH, wp1t, nullptr, Wp2, ssum + 0,
                                                    bp1, 1, 0, N);
        han_wmma_gemm<<<blocks, 256, shm, stream>>>(z1b, HAN_DH, wp1t, nullptr, Wp2, ssum + 1,
                                                    bp1, 1, 0, N);
    }

    // 7) softmax over the two path means
    han_beta<<<1, 1, 0, stream>>>(ssum, beta, 1.0f / (float)N);

    // 8) beta-combine + prediction head
    {
        int blocks = (N * 32 + 255) / 256;
        han_final<<<blocks, 256, 0, stream>>>(z0, z1, beta, Wo, bo, out, N);
    }
    (void)in_sizes; (void)n_in; (void)out_size; (void)ws_size;
}
